// PointNetSetAbstraction_23270132810091
// MI455X (gfx1250) — compile-verified, matrix operand order checked by reading, not run on hardware
//
#include <hip/hip_runtime.h>
#include <hip/hip_bf16.h>

#define BB 16
#define NN 4096
#define DD 64
#define SS 1024
#define KK 32
#define RAD2 (0.4f * 0.4f)
#define EPSN 1e-5f

typedef __attribute__((ext_vector_type(16))) _Float16 v16h;
typedef __attribute__((ext_vector_type(8)))  float    v8f;

struct MlpParams {
    const float* w[3];
    const float* b[3];
    const float* g[3];
    const float* bt[3];
    const float* m[3];
    const float* v[3];
};

// Prepped-weight layout (halves), per layer: [ntiles*ktiles][32 lanes][16]
#define L1_NT 4
#define L1_KT 3
#define L2_NT 4
#define L2_KT 2
#define L3_NT 8
#define L3_KT 2
#define L1_OFF 0
#define L2_OFF (L1_NT * L1_KT * 512)              // 6144 halves
#define L3_OFF (L2_OFF + L2_NT * L2_KT * 512)     // 10240 halves
#define WP_HALVES (L3_OFF + L3_NT * L3_KT * 512)  // 18432 halves

// ---------------------------------------------------------------------------
// Kernel 0: weight / BN prep. Converts weights to f16 in the exact per-lane
// B-operand striping (lane = col | k-half), zero-padded 67->96 for layer 1,
// and folds BN+bias into per-channel (scale, shift) float2s.
// ---------------------------------------------------------------------------
__global__ void prep_kernel(MlpParams P, _Float16* __restrict__ wp,
                            float2* __restrict__ scsh) {
    int t = blockIdx.x * blockDim.x + threadIdx.x;
    if (t < 1152) {                               // 384 + 256 + 512 tile-lane tasks
        int l, rem, KT, cinA, woff;
        if (t < 384)      { l = 0; rem = t;       KT = L1_KT; cinA = 67; woff = L1_OFF; }
        else if (t < 640) { l = 1; rem = t - 384; KT = L2_KT; cinA = 64; woff = L2_OFF; }
        else              { l = 2; rem = t - 640; KT = L3_KT; cinA = 64; woff = L3_OFF; }
        int lane = rem & 31;
        int tile = rem >> 5;                      // tile = nt*KT + kt
        int nt = tile / KT, kt = tile - nt * KT;
        int o  = nt * 16 + (lane & 15);
        int kb = kt * 32 + ((lane >> 4) << 4);
        const float* w = P.w[l];
        _Float16* dst = wp + woff + ((size_t)tile * 32 + lane) * 16;
#pragma unroll
        for (int j = 0; j < 16; ++j) {
            int c = kb + j;
            dst[j] = (c < cinA) ? (_Float16)w[o * cinA + c] : (_Float16)0.0f;
        }
    } else if (t < 1408) {
        int ch = t - 1152;                        // 64 + 64 + 128 channels
        int l, c;
        if (ch < 64)       { l = 0; c = ch; }
        else if (ch < 128) { l = 1; c = ch - 64; }
        else               { l = 2; c = ch - 128; }
        float sc = P.g[l][c] * rsqrtf(P.v[l][c] + EPSN);
        float sh = sc * (P.b[l][c] - P.m[l][c]) + P.bt[l][c];
        scsh[ch] = make_float2(sc, sh);
    }
}

// ---------------------------------------------------------------------------
// Kernel 1: farthest point sampling. One block per batch, 1024 threads (32
// waves). Each thread owns 4 points (registers). Per step: update running
// min-distance, then block-wide argmax via wave32 shuffles + LDS.
// ---------------------------------------------------------------------------
__global__ __launch_bounds__(1024) void fps_kernel(const float* __restrict__ xyz,
                                                   int* __restrict__ fps_idx,
                                                   float* __restrict__ new_xyz) {
    const int b = blockIdx.x;
    const int t = threadIdx.x;
    const float* pts = xyz + (size_t)b * NN * 3;

    float px[4], py[4], pz[4], dist[4];
#pragma unroll
    for (int i = 0; i < 4; ++i) {
        int n = i * 1024 + t;
        px[i] = pts[n * 3 + 0];
        py[i] = pts[n * 3 + 1];
        pz[i] = pts[n * 3 + 2];
        dist[i] = 1e10f;
    }

    __shared__ float sd[32];
    __shared__ int   si[32];
    __shared__ float sC[3];
    __shared__ int   sFar;

    int farthest = 0;
    const int lane = t & 31;
    const int wid  = t >> 5;

    for (int s = 0; s < SS; ++s) {
        if (t == 0) {
            fps_idx[b * SS + s] = farthest;
            float cx = pts[farthest * 3 + 0];
            float cy = pts[farthest * 3 + 1];
            float cz = pts[farthest * 3 + 2];
            sC[0] = cx; sC[1] = cy; sC[2] = cz;
            float* nx = new_xyz + ((size_t)b * SS + s) * 3;
            nx[0] = cx; nx[1] = cy; nx[2] = cz;
        }
        __syncthreads();
        float cx = sC[0], cy = sC[1], cz = sC[2];

        float best = -1.0f; int bi = 0;
#pragma unroll
        for (int i = 0; i < 4; ++i) {
            float dx = px[i] - cx, dy = py[i] - cy, dz = pz[i] - cz;
            float d = dx * dx + dy * dy + dz * dz;
            d = fminf(dist[i], d);
            dist[i] = d;
            int n = i * 1024 + t;
            if (d > best || (d == best && n < bi)) { best = d; bi = n; }
        }
        for (int off = 16; off > 0; off >>= 1) {
            float od = __shfl_xor(best, off);
            int   oi = __shfl_xor(bi,   off);
            if (od > best || (od == best && oi < bi)) { best = od; bi = oi; }
        }
        if (lane == 0) { sd[wid] = best; si[wid] = bi; }
        __syncthreads();
        if (t < 32) {
            float wb = sd[t]; int wi = si[t];
            for (int off = 16; off > 0; off >>= 1) {
                float od = __shfl_xor(wb, off);
                int   oi = __shfl_xor(wi, off);
                if (od > wb || (od == wb && oi < wi)) { wb = od; wi = oi; }
            }
            if (t == 0) sFar = wi;
        }
        __syncthreads();
        farthest = sFar;
    }
}

// ---------------------------------------------------------------------------
// Kernel 2: ball query. One wave32 per center; ballot+popcount ordered
// compaction preserves ascending-index selection of the first K in-radius
// points; shuffle argmin provides the fallback index.
// ---------------------------------------------------------------------------
__global__ __launch_bounds__(256) void ball_query_kernel(const float* __restrict__ xyz,
                                                         const float* __restrict__ new_xyz,
                                                         int* __restrict__ idx_out) {
    const int gid  = blockIdx.x * blockDim.x + threadIdx.x;
    const int wave = gid >> 5;
    const int lane = threadIdx.x & 31;
    if (wave >= BB * SS) return;
    const int b = wave / SS;
    const float* pts = xyz + (size_t)b * NN * 3;
    const float cx = new_xyz[wave * 3 + 0];
    const float cy = new_xyz[wave * 3 + 1];
    const float cz = new_xyz[wave * 3 + 2];
    int* out = idx_out + (size_t)wave * KK;

    int filled = 0;
    float bestd = 1e30f; int besti = 0;
    for (int n0 = 0; n0 < NN; n0 += 32) {
        int n = n0 + lane;
        float dx = pts[n * 3 + 0] - cx;
        float dy = pts[n * 3 + 1] - cy;
        float dz = pts[n * 3 + 2] - cz;
        float d = dx * dx + dy * dy + dz * dz;
        bool inr = d <= RAD2;
        unsigned mask = (unsigned)__ballot(inr);   // wave32: low 32 bits
        if (d < bestd) { bestd = d; besti = n; }
        int prefix = __popc(mask & ((1u << lane) - 1u));
        if (inr) {
            int slot = filled + prefix;
            if (slot < KK) out[slot] = n;
        }
        filled += __popc(mask);
        if (filled >= KK) break;                   // uniform across wave
    }
    if (filled < KK) {
        for (int off = 16; off > 0; off >>= 1) {
            float od = __shfl_xor(bestd, off);
            int   oi = __shfl_xor(besti, off);
            if (od < bestd || (od == bestd && oi < besti)) { bestd = od; besti = oi; }
        }
        for (int k = filled + lane; k < KK; k += 32) out[k] = besti;
    }
}

// ---------------------------------------------------------------------------
// Kernel 3: gather + 3-layer MLP via WMMA + fused BN/ReLU + register max.
// One wave32 per center; 4 waves/block. B-tiles are single v16h loads from
// the prepped f16 weight image (L2-resident). Layer 3 never touches LDS:
// the neighbor-max folds inside the C/D register layout + one shfl_xor(16).
// ---------------------------------------------------------------------------
#define XSTRIDE 96
#define ASTRIDE 64

__device__ __forceinline__ v16h load_a_tile(const _Float16* base, int stride,
                                            int mbase, int kbase, int lane) {
    int row  = mbase + (lane & 15);
    int kofs = kbase + ((lane >> 4) << 4);
    return *(const v16h*)(base + row * stride + kofs);   // 32B-aligned LDS read
}

template <int NT, int KT>
__device__ __forceinline__ void mlp_layer_ws(const _Float16* inL, int instride,
                                             const _Float16* __restrict__ wp,
                                             const float2* __restrict__ scsh,
                                             _Float16* outL, int outstride,
                                             int lane) {
#pragma unroll
    for (int mt = 0; mt < 2; ++mt) {
        v16h a[KT];
#pragma unroll
        for (int kt = 0; kt < KT; ++kt)
            a[kt] = load_a_tile(inL, instride, mt * 16, kt * 32, lane);
#pragma unroll
        for (int nt = 0; nt < NT; ++nt) {
            v8f c = {};
#pragma unroll
            for (int kt = 0; kt < KT; ++kt) {
                v16h bm = *(const v16h*)(wp + ((size_t)(nt * KT + kt) * 32 + lane) * 16);
                c = __builtin_amdgcn_wmma_f32_16x16x32_f16(
                        false, a[kt], false, bm, (short)0, c, false, false);
            }
            float2 ss = scsh[nt * 16 + (lane & 15)];
            int col   = nt * 16 + (lane & 15);
            int rbase = mt * 16 + ((lane >> 4) << 3);
#pragma unroll
            for (int r = 0; r < 8; ++r) {
                float y = fmaxf(ss.x * c[r] + ss.y, 0.0f);
                outL[(rbase + r) * outstride + col] = (_Float16)y;
            }
        }
    }
}

template <int NT, int KT>
__device__ __forceinline__ void mlp_layer_max(const _Float16* inL, int instride,
                                              const _Float16* __restrict__ wp,
                                              const float2* __restrict__ scsh,
                                              float* __restrict__ ofe, int lane) {
    v16h a0[KT], a1[KT];
#pragma unroll
    for (int kt = 0; kt < KT; ++kt) {
        a0[kt] = load_a_tile(inL, instride, 0,  kt * 32, lane);
        a1[kt] = load_a_tile(inL, instride, 16, kt * 32, lane);
    }
#pragma unroll
    for (int nt = 0; nt < NT; ++nt) {
        v8f c0 = {}, c1 = {};
#pragma unroll
        for (int kt = 0; kt < KT; ++kt) {
            v16h bm = *(const v16h*)(wp + ((size_t)(nt * KT + kt) * 32 + lane) * 16);
            c0 = __builtin_amdgcn_wmma_f32_16x16x32_f16(
                     false, a0[kt], false, bm, (short)0, c0, false, false);
            c1 = __builtin_amdgcn_wmma_f32_16x16x32_f16(
                     false, a1[kt], false, bm, (short)0, c1, false, false);
        }
        float2 ss = scsh[nt * 16 + (lane & 15)];
        float mx = 0.0f;                       // ReLU outputs are >= 0
#pragma unroll
        for (int r = 0; r < 8; ++r) {
            mx = fmaxf(mx, fmaxf(ss.x * c0[r] + ss.y, 0.0f));
            mx = fmaxf(mx, fmaxf(ss.x * c1[r] + ss.y, 0.0f));
        }
        // lane l and l^16 hold the same column with complementary rows
        mx = fmaxf(mx, __shfl_xor(mx, 16));
        if (lane < 16) ofe[nt * 16 + lane] = mx;
    }
}

__global__ __launch_bounds__(128) void group_mlp_kernel(const float* __restrict__ xyz,
                                                        const float* __restrict__ feat,
                                                        const float* __restrict__ new_xyz,
                                                        const int* __restrict__ ball_idx,
                                                        const _Float16* __restrict__ wp,
                                                        const float2* __restrict__ scsh,
                                                        float* __restrict__ out_feat) {
    __shared__ __align__(32) _Float16 sX[4][32 * XSTRIDE];   // 24 KB
    __shared__ __align__(32) _Float16 sA[4][32 * ASTRIDE];   // 16 KB

    const int lane   = threadIdx.x & 31;
    const int wv     = threadIdx.x >> 5;
    const int center = blockIdx.x * 4 + wv;
    const int b = center / SS;

    // ---- gather: lane owns neighbor row `lane` of the 32 x 96 tile --------
    const int n = ball_idx[(size_t)center * KK + lane];
    const float* p = xyz + ((size_t)b * NN + n) * 3;
    const float* f = feat + ((size_t)b * NN + n) * DD;
    __builtin_prefetch(f, 0, 3);                         // global_prefetch_b8
    const float cx = new_xyz[center * 3 + 0];
    const float cy = new_xyz[center * 3 + 1];
    const float cz = new_xyz[center * 3 + 2];

    _Float16* X = &sX[wv][lane * XSTRIDE];
    X[0] = (_Float16)(p[0] - cx);
    X[1] = (_Float16)(p[1] - cy);
    X[2] = (_Float16)(p[2] - cz);
#pragma unroll 4
    for (int j = 0; j < DD; j += 4) {
        float4 fv = *(const float4*)(f + j);
        X[3 + j + 0] = (_Float16)fv.x;
        X[3 + j + 1] = (_Float16)fv.y;
        X[3 + j + 2] = (_Float16)fv.z;
        X[3 + j + 3] = (_Float16)fv.w;
    }
#pragma unroll
    for (int j = 3 + DD; j < XSTRIDE; ++j) X[j] = (_Float16)0.0f;

    // ---- MLP stack --------------------------------------------------------
    _Float16* Xb = sX[wv];
    _Float16* A  = sA[wv];
    mlp_layer_ws<L1_NT, L1_KT>(Xb, XSTRIDE, wp + L1_OFF, scsh,       A,  ASTRIDE, lane);
    mlp_layer_ws<L2_NT, L2_KT>(A,  ASTRIDE, wp + L2_OFF, scsh + 64,  Xb, ASTRIDE, lane);
    mlp_layer_max<L3_NT, L3_KT>(Xb, ASTRIDE, wp + L3_OFF, scsh + 128,
                                out_feat + (size_t)center * 128, lane);
}

// ---------------------------------------------------------------------------
extern "C" void kernel_launch(void* const* d_in, const int* in_sizes, int n_in,
                              void* d_out, int out_size, void* d_ws, size_t ws_size,
                              hipStream_t stream) {
    const float* xyz  = (const float*)d_in[0];
    const float* feat = (const float*)d_in[1];

    MlpParams P;
    for (int i = 0; i < 3; ++i) {
        P.w[i]  = (const float*)d_in[2 + 6 * i + 0];
        P.b[i]  = (const float*)d_in[2 + 6 * i + 1];
        P.g[i]  = (const float*)d_in[2 + 6 * i + 2];
        P.bt[i] = (const float*)d_in[2 + 6 * i + 3];
        P.m[i]  = (const float*)d_in[2 + 6 * i + 4];
        P.v[i]  = (const float*)d_in[2 + 6 * i + 5];
    }

    float* out      = (float*)d_out;
    float* new_xyz  = out;                                 // [B,S,3]
    float* new_feat = out + (size_t)BB * SS * 3;           // [B,S,128]
    int*   fps_out  = (int*)(out + (size_t)BB * SS * 3 + (size_t)BB * SS * 128);

    // Workspace: ball indices | prepped f16 weights | (scale,shift) pairs
    char* ws = (char*)d_ws;
    int*       ball  = (int*)ws;                                       // 2 MB
    _Float16*  wp    = (_Float16*)(ws + (size_t)BB * SS * KK * 4);     // 36 KB
    float2*    scsh  = (float2*)(ws + (size_t)BB * SS * KK * 4
                                    + (size_t)WP_HALVES * 2);          // 2 KB

    prep_kernel<<<6, 256, 0, stream>>>(P, wp, scsh);

    fps_kernel<<<BB, 1024, 0, stream>>>(xyz, fps_out, new_xyz);

    {
        int waves = BB * SS;
        int threads = 256;
        int blocks = (waves * 32 + threads - 1) / threads;
        ball_query_kernel<<<blocks, threads, 0, stream>>>(xyz, new_xyz, ball);
    }

    group_mlp_kernel<<<(BB * SS) / 4, 128, 0, stream>>>(xyz, feat, new_xyz, ball,
                                                        wp, scsh, new_feat);
}
